// EncoderDecoder_40707700032197
// MI455X (gfx1250) — matrix operand order repaired and checked by reading, not compiled, not run on hardware
//
#include <hip/hip_runtime.h>

// CDNA5 / gfx1250 encoder-decoder LSTM.
// Encoder: 288 recurrent steps; z = [h | x | 1] @ [Whh ; Wih ; b]^T done
//          entirely with v_wmma_f32_16x16x32_f16 (5 K-chunks: 4 for h,
//          1 for the 7-wide input + fused bias). All B fragments are
//          register-resident for the whole kernel. The [x|1] A-fragment is
//          built branch-free from a 32-row staging buffer whose upper half
//          is permanently zero (lanes 16-31 need zeros per the A layout).
//          Gate nonlinearities are branch-free (v_tanh_f32 / v_exp + v_rcp).
// Decoder: exploits state-reset semantics -> one WMMA GEMM for z_base, then
//          288 cheap elementwise steps (3-wide feedback only).

typedef __attribute__((ext_vector_type(16))) _Float16 v16h;
typedef __attribute__((ext_vector_type(8)))  float    v8f;

#define B_ALL   1024
#define SEQ_T   288
#define TGT_T   288
#define INP     7
#define HID     128
#define G4H     512     // 4*HID
#define OUTN    3
#define MTILE   16      // batch rows per workgroup
#define H16S    144     // padded f16 row stride (byte stride 288 = 16-aligned at kb*2)

// Branch-free transcendentals for the serialized gate phase.
#if defined(__has_builtin)
#  if __has_builtin(__builtin_amdgcn_tanhf)
#    define TANH_F32(x) __builtin_amdgcn_tanhf(x)
#  elif __has_builtin(__builtin_amdgcn_tanh_f32)
#    define TANH_F32(x) __builtin_amdgcn_tanh_f32(x)
#  endif
#endif
#ifndef TANH_F32
   // exact identity, lowers to v_exp_f32 + v_rcp_f32, no branches
#  define TANH_F32(x) (1.0f - 2.0f * __builtin_amdgcn_rcpf(__expf(2.0f * (x)) + 1.0f))
#endif

__device__ __forceinline__ float sigf(float x) {
    return __builtin_amdgcn_rcpf(1.0f + __expf(-x));   // v_exp + v_rcp, branch-free
}

__global__ __launch_bounds__(256, 1) void lstm_encdec_kernel(
    const float* __restrict__ x,     // (1024, 288, 7)
    const float* __restrict__ eWih,  // (512, 7)
    const float* __restrict__ eWhh,  // (512, 128)
    const float* __restrict__ eBih,  // (512)
    const float* __restrict__ eBhh,  // (512)
    const float* __restrict__ dWih,  // (512, 3)
    const float* __restrict__ dWhh,  // (512, 128)
    const float* __restrict__ dBih,  // (512)
    const float* __restrict__ dBhh,  // (512)
    const float* __restrict__ oW,    // (3, 128)
    const float* __restrict__ oB,    // (3)
    float* __restrict__ out)         // (1024, 288, 3)
{
    __shared__ _Float16 h16[MTILE * H16S];   // h state, f16 (WMMA A source)
    __shared__ float    zs[MTILE * G4H];     // z scratch / decoder z_base
    __shared__ float    cs[MTILE * HID];     // c state (f32)
    __shared__ float    hdec[MTILE * HID];   // decoder h (f32)
    __shared__ float    xs[32 * 8];          // staged [x|1]; rows 16-31 stay zero
    __shared__ float    ys[MTILE * 4];       // decoder feedback y_prev
    __shared__ float    dwihS[G4H * 3];      // dec_Wih
    __shared__ float    owS[OUTN * HID];     // out_W
    __shared__ float    obS[4];              // out_b

    const int tid    = threadIdx.x;
    const int wave   = tid >> 5;             // 0..7, owns N cols [wave*64, wave*64+64)
    const int lane   = tid & 31;
    const int r0     = blockIdx.x * MTILE;

    const int ncl    = lane & 15;            // local column within a 16-wide N tile
    const int hiHalf = (lane >> 4) & 1;      // lane group 0/1
    const int mbase  = hiHalf * 8;           // C/D fragment row base (ISA 7.12.2)

    // ---- zero-init h, c; set up the constant parts of xs ----
    for (int i = tid; i < MTILE * H16S; i += 256) h16[i] = (_Float16)0.0f;
    for (int i = tid; i < MTILE * HID;  i += 256) cs[i]  = 0.0f;
    if (tid < 128) xs[128 + tid] = 0.0f;            // rows 16..31 = 0 forever
    if (tid < 16)  xs[tid * 8 + 7] = 1.0f;          // bias multiplier column

    // ---- register-resident B fragments ----
    // B (32x16, f16): lanes 0-15 hold K=kb..kb+15 for N=lane&15,
    //                 lanes 16-31 hold K=kb+16..kb+31.
    v16h bf[4][4];   // [ntile][kchunk] : Whh,   K = kc*32 ..
    v16h bfx[4];     // [ntile]         : Wih rows 0..6, fused bias at row 7, rest 0
    #pragma unroll
    for (int tn = 0; tn < 4; ++tn) {
        const int nn = wave * 64 + tn * 16 + ncl;
        #pragma unroll
        for (int kc = 0; kc < 4; ++kc) {
            const int kb = kc * 32 + hiHalf * 16;
            const float* wp = &eWhh[nn * HID + kb];   // B[k][n] = Whh[n][k]
            #pragma unroll
            for (int j = 0; j < 16; ++j) bf[tn][kc][j] = (_Float16)wp[j];
        }
        #pragma unroll
        for (int j = 0; j < 16; ++j) bfx[tn][j] = (_Float16)0.0f;
        if (!hiHalf) {
            #pragma unroll
            for (int j = 0; j < INP; ++j) bfx[tn][j] = (_Float16)eWih[nn * INP + j];
            bfx[tn][7] = (_Float16)(eBih[nn] + eBhh[nn]);   // bias row, pairs with A==1
        }
    }
    __syncthreads();

    // ================= encoder: 288 recurrent steps =================
    for (int t = 0; t < SEQ_T; ++t) {
        // stage x_t (16 rows x 7) into the live part of xs
        if (tid < MTILE * INP) {
            const int m = tid / INP, k = tid % INP;
            xs[m * 8 + k] = x[((size_t)(r0 + m) * SEQ_T + t) * INP + k];
        }
        __syncthreads();

        // A fragment for the [x | 1] K-chunk, branch-free:
        // lanes 0-15 read live rows (K=0..7), lanes 16-31 read the zero rows.
        v16h ax;
        {
            const float* xr = &xs[(hiHalf * 16 + ncl) * 8];
            #pragma unroll
            for (int j = 0; j < 8; ++j) ax[j] = (_Float16)xr[j];
            #pragma unroll
            for (int j = 8; j < 16; ++j) ax[j] = (_Float16)0.0f;  // K=16..23 > 7
        }

        v8f acc[4];
        #pragma unroll
        for (int tn = 0; tn < 4; ++tn) {
            #pragma unroll
            for (int v = 0; v < 8; ++v) acc[tn][v] = 0.0f;
            acc[tn] = __builtin_amdgcn_wmma_f32_16x16x32_f16(
                false, ax, false, bfx[tn], (short)0, acc[tn], false, false);
        }

        // h @ Whh^T via WMMA: 4 K-chunks x 4 N-tiles
        #pragma unroll
        for (int kc = 0; kc < 4; ++kc) {
            // A (16x32, f16): elements 0..15 -> K=kb..kb+15 (contiguous, kb includes hiHalf*16)
            v16h a;
            const int kb = kc * 32 + hiHalf * 16;
            const _Float16* hp = &h16[ncl * H16S + kb];
            #pragma unroll
            for (int j = 0; j < 8; ++j) { a[j] = hp[j]; a[8 + j] = hp[8 + j]; }
            #pragma unroll
            for (int tn = 0; tn < 4; ++tn)
                acc[tn] = __builtin_amdgcn_wmma_f32_16x16x32_f16(
                    false, a, false, bf[tn][kc], (short)0, acc[tn], false, false);
        }

        // spill z (C/D layout: VGPR v -> row mbase+v, lane -> col)
        #pragma unroll
        for (int tn = 0; tn < 4; ++tn) {
            const int nc = wave * 64 + tn * 16 + ncl;
            #pragma unroll
            for (int v = 0; v < 8; ++v)
                zs[(mbase + v) * G4H + nc] = acc[tn][v];
        }
        __syncthreads();

        // fused gate phase: each thread owns (b, 8 consecutive h columns)
        {
            const int b   = tid >> 4;
            const int hc0 = (tid & 15) * 8;
            #pragma unroll
            for (int u = 0; u < 8; ++u) {
                const int hc = hc0 + u;
                const float zi = zs[b * G4H + hc];
                const float zf = zs[b * G4H + HID + hc];
                const float zg = zs[b * G4H + 2 * HID + hc];
                const float zo = zs[b * G4H + 3 * HID + hc];
                const float cO = cs[b * HID + hc];
                const float cn = sigf(zf) * cO + sigf(zi) * TANH_F32(zg);
                const float hn = sigf(zo) * TANH_F32(cn);
                cs[b * HID + hc]   = cn;
                h16[b * H16S + hc] = (_Float16)hn;
            }
        }
        // barrier at top of next iteration orders gates(t) vs GEMM(t+1)
    }
    __syncthreads();   // h16 = h_enc (f16), cs = c_enc

    // ================= decoder setup =================
    // swap register B fragments to decoder Whh
    float bias_r[4];
    #pragma unroll
    for (int tn = 0; tn < 4; ++tn) {
        const int nn = wave * 64 + tn * 16 + ncl;
        bias_r[tn] = dBih[nn] + dBhh[nn];
        #pragma unroll
        for (int kc = 0; kc < 4; ++kc) {
            const int kb = kc * 32 + hiHalf * 16;
            const float* wp = &dWhh[nn * HID + kb];
            #pragma unroll
            for (int j = 0; j < 16; ++j) bf[tn][kc][j] = (_Float16)wp[j];
        }
    }
    // stage small decoder weights
    for (int i = tid; i < G4H * 3;    i += 256) dwihS[i] = dWih[i];
    for (int i = tid; i < OUTN * HID; i += 256) owS[i]   = oW[i];
    if (tid < OUTN)      obS[tid] = oB[tid];
    if (tid < MTILE * 4) ys[tid]  = 0.0f;      // y_prev = 0

    // z_base = h_enc @ dec_Whh^T + b_dec  (single WMMA GEMM — state resets!)
    {
        v8f acc[4];
        #pragma unroll
        for (int tn = 0; tn < 4; ++tn)
            #pragma unroll
            for (int v = 0; v < 8; ++v) acc[tn][v] = bias_r[tn];
        #pragma unroll
        for (int kc = 0; kc < 4; ++kc) {
            v16h a;
            const int kb = kc * 32 + hiHalf * 16;
            const _Float16* hp = &h16[ncl * H16S + kb];
            #pragma unroll
            for (int j = 0; j < 8; ++j) { a[j] = hp[j]; a[8 + j] = hp[8 + j]; }
            #pragma unroll
            for (int tn = 0; tn < 4; ++tn)
                acc[tn] = __builtin_amdgcn_wmma_f32_16x16x32_f16(
                    false, a, false, bf[tn][kc], (short)0, acc[tn], false, false);
        }
        #pragma unroll
        for (int tn = 0; tn < 4; ++tn) {
            const int nc = wave * 64 + tn * 16 + ncl;
            #pragma unroll
            for (int v = 0; v < 8; ++v)
                zs[(mbase + v) * G4H + nc] = acc[tn][v];   // persistent z_base
        }
    }
    __syncthreads();

    // ================= decoder: 288 cheap steps =================
    const int b_g   = tid >> 4;
    const int hc0_g = (tid & 15) * 8;
    for (int t = 0; t < TGT_T; ++t) {
        const float y0v = ys[b_g * 4 + 0];
        const float y1v = ys[b_g * 4 + 1];
        const float y2v = ys[b_g * 4 + 2];
        #pragma unroll
        for (int u = 0; u < 8; ++u) {
            const int hc = hc0_g + u;
            float zg4[4];
            #pragma unroll
            for (int g = 0; g < 4; ++g) {
                const int n = g * HID + hc;
                zg4[g] = zs[b_g * G4H + n]
                       + dwihS[n * 3 + 0] * y0v
                       + dwihS[n * 3 + 1] * y1v
                       + dwihS[n * 3 + 2] * y2v;
            }
            const float cO = cs[b_g * HID + hc];         // c_enc, read-only
            const float cn = sigf(zg4[1]) * cO + sigf(zg4[0]) * TANH_F32(zg4[2]);
            const float hn = sigf(zg4[3]) * TANH_F32(cn);
            hdec[b_g * HID + hc] = hn;
        }
        __syncthreads();

        // y = h @ out_W^T + out_b ; write output and feedback
        if (tid < MTILE * OUTN) {
            const int b = tid / OUTN, o = tid % OUTN;
            float s = obS[o];
            const float* hp = &hdec[b * HID];
            const float* wp = &owS[o * HID];
            #pragma unroll 8
            for (int k = 0; k < HID; ++k) s += hp[k] * wp[k];
            out[((size_t)(r0 + b) * TGT_T + t) * OUTN + o] = s;
            ys[b * 4 + o] = s;
        }
        __syncthreads();
    }
}

extern "C" void kernel_launch(void* const* d_in, const int* in_sizes, int n_in,
                              void* d_out, int out_size, void* d_ws, size_t ws_size,
                              hipStream_t stream) {
    (void)in_sizes; (void)n_in; (void)out_size; (void)d_ws; (void)ws_size;
    lstm_encdec_kernel<<<B_ALL / MTILE, 256, 0, stream>>>(
        (const float*)d_in[0],   // x
        (const float*)d_in[1],   // enc_Wih
        (const float*)d_in[2],   // enc_Whh
        (const float*)d_in[3],   // enc_bih
        (const float*)d_in[4],   // enc_bhh
        (const float*)d_in[5],   // dec_Wih
        (const float*)d_in[6],   // dec_Whh
        (const float*)d_in[7],   // dec_bih
        (const float*)d_in[8],   // dec_bhh
        (const float*)d_in[9],   // out_W
        (const float*)d_in[10],  // out_b
        (float*)d_out);
}
